// DAGNN2021Encoder_16947940950533
// MI455X (gfx1250) — matrix operand order repaired and checked by reading, not compiled, not run on hardware
//
#include <hip/hip_runtime.h>
#include <hip/hip_bf16.h>

typedef __attribute__((ext_vector_type(16))) _Float16 v16h;
typedef __attribute__((ext_vector_type(8)))  float    v8f;

struct alignas(16) H8 { _Float16 h[8]; };

// ---------------------------------------------------------------------------
// Fragment loaders for v_wmma_f32_16x16x32_f16
// A 16x32 (f16): lane r=lane&15 is row M; hi=lane>>4.
//   halves[0..7]  = K = hi*8 + (0..7)
//   halves[8..15] = K = 16 + hi*8 + (0..7)
// B 32x16 (f16): lane n=lane&15 is col N; same K order as A:
//   halves[0..7] = K = hi*8 + (0..7), halves[8..15] = K = 16 + hi*8 + (0..7).
// ---------------------------------------------------------------------------
__device__ __forceinline__ v16h load_fragA(const _Float16* row_ptr, int k0, int hi) {
  H8 lo = *reinterpret_cast<const H8*>(row_ptr + k0 + hi * 8);
  H8 hv = *reinterpret_cast<const H8*>(row_ptr + k0 + 16 + hi * 8);
  v16h a;
#pragma unroll
  for (int i = 0; i < 8; ++i) { a[i] = lo.h[i]; a[8 + i] = hv.h[i]; }
  return a;
}

__device__ __forceinline__ v16h load_fragB(const _Float16* row_ptr, int k0, int hi) {
  H8 b0 = *reinterpret_cast<const H8*>(row_ptr + k0 + hi * 8);
  H8 b1 = *reinterpret_cast<const H8*>(row_ptr + k0 + 16 + hi * 8);
  v16h b;
#pragma unroll
  for (int i = 0; i < 8; ++i) { b[i] = b0.h[i]; b[8 + i] = b1.h[i]; }
  return b;
}

// ---------------------------------------------------------------------------
// WMMA GEMM: C = A @ B^T + bias. All configuration is compile-time so the
// hot loop is branch-free straight-line code (no accumulator copies).
// Block = 128 threads (4 waves); block tile = 64 rows x 64 cols; each wave
// owns a 16-row stripe and 4 column tiles. K fully unrolled.
// ---------------------------------------------------------------------------
template <int KDIM, bool DUAL, bool GATHER, bool OUTH, bool OUTF>
__global__ void __launch_bounds__(128)
gemm_wmma_kernel(const _Float16* __restrict__ A,
                 const int* __restrict__ row_idx,
                 const _Float16* __restrict__ B0, const float* __restrict__ bias0,
                 const _Float16* __restrict__ B1, const float* __restrict__ bias1,
                 _Float16* __restrict__ C0h, int ldc0h,
                 float*    __restrict__ C0f, int ldc0f,
                 _Float16* __restrict__ C1h, int ldc1h)
{
  const int lane = threadIdx.x & 31;
  const int wave = threadIdx.x >> 5;
  const int r    = lane & 15;
  const int hi   = lane >> 4;
  const int rowTile = blockIdx.y * 64 + wave * 16;
  const int colBase = blockIdx.x * 64;

  const int arow = rowTile + r;
  const _Float16* arow_ptr;
  if (GATHER) arow_ptr = A + (size_t)row_idx[arow] * KDIM;
  else        arow_ptr = A + (size_t)arow * KDIM;

  const _Float16* b0base = B0 + (size_t)(colBase + r) * KDIM;
  const _Float16* b1base = DUAL ? (B1 + (size_t)(colBase + r) * KDIM) : nullptr;

  v8f acc0[4] = {};
  v8f acc1[4] = {};

#pragma unroll
  for (int k0 = 0; k0 < KDIM; k0 += 32) {
    v16h a = load_fragA(arow_ptr, k0, hi);
#pragma unroll
    for (int t = 0; t < 4; ++t) {
      v16h b = load_fragB(b0base + (size_t)t * 16 * KDIM, k0, hi);
      acc0[t] = __builtin_amdgcn_wmma_f32_16x16x32_f16(
          false, a, false, b, (short)0, acc0[t], false, false);
      if (DUAL) {
        v16h b1 = load_fragB(b1base + (size_t)t * 16 * KDIM, k0, hi);
        acc1[t] = __builtin_amdgcn_wmma_f32_16x16x32_f16(
            false, a, false, b1, (short)0, acc1[t], false, false);
      }
    }
  }

#pragma unroll
  for (int t = 0; t < 4; ++t) {
    const int n = colBase + t * 16 + r;   // column held by this lane
    const float bv0 = bias0[n];
#pragma unroll
    for (int v = 0; v < 8; ++v) {
      const int m = rowTile + hi * 8 + v;
      const float val = acc0[t][v] + bv0;
      if (OUTH) C0h[(size_t)m * ldc0h + n] = (_Float16)val;
      if (OUTF) C0f[(size_t)m * ldc0f + n] = val;
    }
    if (DUAL) {
      const float bv1 = bias1[n];
#pragma unroll
      for (int v = 0; v < 8; ++v) {
        const int m = rowTile + hi * 8 + v;
        C1h[(size_t)m * ldc1h + n] = (_Float16)(acc1[t][v] + bv1);
      }
    }
  }
}

// ---------------------------------------------------------------------------
// Attention: one wave per node. 4 heads, HD=64, P=16 preds.
// Lane -> (p = lane&15, head-pair = lane>>4). Softmax over the 16-lane group
// via xor shuffles (masks < 16 stay within the half-wave). Aggregation
// redistributed through LDS: each lane produces 8 of the 256 agg channels.
// Also copies the q row (prev state) into the first half of the concat buffer.
// ---------------------------------------------------------------------------
__global__ void __launch_bounds__(256)
attn_kernel(const _Float16* __restrict__ Q16,   // [4096,256]
            const _Float16* __restrict__ K16,   // [65536,256]
            const _Float16* __restrict__ V16,   // [65536,256]
            const _Float16* __restrict__ prevQ, // prev16 + qbase*256
            _Float16* __restrict__ agg16,       // [4096,256]
            _Float16* __restrict__ cat16)       // [4096,512]
{
  __shared__ float sattn[8][4][16];
  const int wave = threadIdx.x >> 5;
  const int lane = threadIdx.x & 31;
  const int node = blockIdx.x * 8 + wave;
  const int p    = lane & 15;
  const int hh   = lane >> 4;
  const float scale = 0.125f;  // 1/sqrt(64)

  const _Float16* qrow = Q16 + (size_t)node * 256;
  const _Float16* krow = K16 + ((size_t)node * 16 + p) * 256;

  float sc[2];
#pragma unroll
  for (int j = 0; j < 2; ++j) {
    const int h = hh * 2 + j;
    float s = 0.0f;
#pragma unroll
    for (int c = 0; c < 8; ++c) {
      H8 qv = *reinterpret_cast<const H8*>(qrow + h * 64 + c * 8);
      H8 kv = *reinterpret_cast<const H8*>(krow + h * 64 + c * 8);
#pragma unroll
      for (int i = 0; i < 8; ++i) s += (float)qv.h[i] * (float)kv.h[i];
    }
    sc[j] = s * scale;
  }

#pragma unroll
  for (int j = 0; j < 2; ++j) {
    float mx = sc[j];
    for (int msk = 8; msk >= 1; msk >>= 1) mx = fmaxf(mx, __shfl_xor(mx, msk, 32));
    const float e = __expf(sc[j] - mx);
    float sm = e;
    for (int msk = 8; msk >= 1; msk >>= 1) sm += __shfl_xor(sm, msk, 32);
    sattn[wave][hh * 2 + j][p] = e / sm;
  }
  __syncthreads();

  float accv[8] = {};
  const int cbase = lane * 8;
  const int h = cbase >> 6;
#pragma unroll
  for (int pp = 0; pp < 16; ++pp) {
    const float a = sattn[wave][h][pp];
    H8 vv = *reinterpret_cast<const H8*>(V16 + ((size_t)node * 16 + pp) * 256 + cbase);
#pragma unroll
    for (int i = 0; i < 8; ++i) accv[i] += a * (float)vv.h[i];
  }
#pragma unroll
  for (int i = 0; i < 8; ++i)
    agg16[(size_t)node * 256 + cbase + i] = (_Float16)accv[i];

  // q -> front half of concat buffer
  H8 qc = *reinterpret_cast<const H8*>(prevQ + (size_t)node * 256 + cbase);
  *reinterpret_cast<H8*>(cat16 + (size_t)node * 512 + cbase) = qc;
}

// ---------------------------------------------------------------------------
// LayerNorm + exact GELU. One wave per 256-wide row; lane owns 8 channels.
// Writes f32 into d_out (row stride 1024) and f16 into the working state.
// ---------------------------------------------------------------------------
__global__ void __launch_bounds__(256)
ln_gelu_kernel(const float* __restrict__ z,      // [4096,256]
               const float* __restrict__ g, const float* __restrict__ be,
               float* __restrict__ outf,         // pre-offset, stride 1024
               _Float16* __restrict__ outh)      // pre-offset, stride 256
{
  const int wave = threadIdx.x >> 5;
  const int lane = threadIdx.x & 31;
  const int row  = blockIdx.x * 8 + wave;
  const float* zr = z + (size_t)row * 256;

  float vals[8];
  float s = 0.0f, ss = 0.0f;
#pragma unroll
  for (int i = 0; i < 8; ++i) {
    const float v = zr[lane * 8 + i];
    vals[i] = v; s += v; ss += v * v;
  }
  for (int msk = 16; msk >= 1; msk >>= 1) {
    s  += __shfl_xor(s,  msk, 32);
    ss += __shfl_xor(ss, msk, 32);
  }
  const float mean = s * (1.0f / 256.0f);
  const float var  = ss * (1.0f / 256.0f) - mean * mean;
  const float inv  = rsqrtf(var + 1e-5f);
#pragma unroll
  for (int i = 0; i < 8; ++i) {
    const int c = lane * 8 + i;
    float v = (vals[i] - mean) * inv * g[c] + be[c];
    const float ge = 0.5f * v * (1.0f + erff(v * 0.70710678118654752f));
    outf[(size_t)row * 1024 + c] = ge;
    outh[(size_t)row * 256 + c] = (_Float16)ge;
  }
}

// ---------------------------------------------------------------------------
// Utility kernels
// ---------------------------------------------------------------------------
__global__ void f32_to_f16_kernel(const float* __restrict__ in,
                                  _Float16* __restrict__ out, size_t n) {
  size_t i = (size_t)blockIdx.x * blockDim.x + threadIdx.x;
  const size_t stride = (size_t)gridDim.x * blockDim.x;
  for (; i < n; i += stride) out[i] = (_Float16)in[i];
}

__global__ void copy_f16_kernel(const _Float16* __restrict__ in,
                                _Float16* __restrict__ out, size_t n) {
  size_t i = (size_t)blockIdx.x * blockDim.x + threadIdx.x;
  const size_t stride = (size_t)gridDim.x * blockDim.x;
  for (; i < n; i += stride) out[i] = in[i];
}

__global__ void copy_lvl0_f32_kernel(const float* __restrict__ in,
                                     float* __restrict__ out) {
  // 4096 rows x 256 cols, both stride 1024
  const int idx = blockIdx.x * blockDim.x + threadIdx.x;  // < 1048576
  const int r = idx >> 8, c = idx & 255;
  out[(size_t)r * 1024 + c] = in[(size_t)r * 1024 + c];
}

// ---------------------------------------------------------------------------
// Host: orchestrate the 45 sequential stages on `stream`.
// ---------------------------------------------------------------------------
extern "C" void kernel_launch(void* const* d_in, const int* in_sizes, int n_in,
                              void* d_out, int out_size, void* d_ws, size_t ws_size,
                              hipStream_t stream) {
  (void)in_sizes; (void)n_in; (void)out_size; (void)ws_size;

  const float* x      = (const float*)d_in[0];
  const int*   preds  = (const int*)  d_in[3];   // [15,4096,16]
  const float* W_in   = (const float*)d_in[4];
  const float* b_in   = (const float*)d_in[5];
  const float* Wq     = (const float*)d_in[6];
  const float* bq     = (const float*)d_in[7];
  const float* Wk     = (const float*)d_in[8];
  const float* bk     = (const float*)d_in[9];
  const float* Wv     = (const float*)d_in[10];
  const float* bv     = (const float*)d_in[11];
  const float* Wo     = (const float*)d_in[12];
  const float* bo     = (const float*)d_in[13];
  const float* Wc     = (const float*)d_in[14];
  const float* bc     = (const float*)d_in[15];
  const float* ln_g   = (const float*)d_in[16];
  const float* ln_b   = (const float*)d_in[17];
  float* out          = (float*)d_out;

  const size_t N  = 65536;   // nodes
  const size_t Mn = 4096;    // nodes per level

  // Workspace carve-out
  char* ws = (char*)d_ws;
  size_t off = 0;
  auto alloc = [&](size_t bytes) -> void* {
    void* p = ws + off;
    off = (off + bytes + 255) & ~(size_t)255;
    return p;
  };
  _Float16* x16   = (_Float16*)alloc(N * 256 * 2);
  _Float16* wIn16 = (_Float16*)alloc(65536 * 2);
  _Float16* wq16  = (_Float16*)alloc(3 * 65536 * 2);
  _Float16* wk16  = (_Float16*)alloc(3 * 65536 * 2);
  _Float16* wv16  = (_Float16*)alloc(3 * 65536 * 2);
  _Float16* wo16  = (_Float16*)alloc(3 * 65536 * 2);
  _Float16* wc16  = (_Float16*)alloc(3 * 131072 * 2);
  _Float16* hA    = (_Float16*)alloc(N * 256 * 2);
  _Float16* hB    = (_Float16*)alloc(N * 256 * 2);
  _Float16* K16   = (_Float16*)alloc(N * 256 * 2);
  _Float16* V16   = (_Float16*)alloc(N * 256 * 2);
  _Float16* Q16   = (_Float16*)alloc(Mn * 256 * 2);
  _Float16* agg16 = (_Float16*)alloc(Mn * 256 * 2);
  _Float16* cat16 = (_Float16*)alloc(Mn * 512 * 2);
  float*    z32   = (float*)   alloc(Mn * 256 * 4);

  auto conv = [&](const float* src, _Float16* dst, size_t n) {
    int blocks = (int)((n + 255 * 8) / (256 * 8));
    if (blocks > 4096) blocks = 4096;
    if (blocks < 1) blocks = 1;
    f32_to_f16_kernel<<<blocks, 256, 0, stream>>>(src, dst, n);
  };

  // 1) f16 mirrors of inputs/weights (weights stay L2-resident: ~2.4 MB)
  conv(x,    x16,   N * 256);
  conv(W_in, wIn16, 65536);
  conv(Wq,   wq16,  3 * 65536);
  conv(Wk,   wk16,  3 * 65536);
  conv(Wv,   wv16,  3 * 65536);
  conv(Wo,   wo16,  3 * 65536);
  conv(Wc,   wc16,  3 * 131072);

  // 2) input projection: h0 = x @ W_in^T + b_in  (f32 -> d_out col 0, f16 -> hA)
  gemm_wmma_kernel<256, false, false, true, true>
      <<<dim3(4, N / 64), 128, 0, stream>>>(
          x16, nullptr, wIn16, b_in, nullptr, nullptr,
          hA, 256, out, 1024, nullptr, 0);

  // 3) three layers of 15 sequential level-steps
  _Float16* prev16 = hA;
  _Float16* cur16  = hB;
  for (int l = 0; l < 3; ++l) {
    // level-0 rows never updated: copy prev into this layer's output slot
    copy_lvl0_f32_kernel<<<4096, 256, 0, stream>>>(out + (size_t)l * 256,
                                                   out + (size_t)(l + 1) * 256);
    copy_f16_kernel<<<512, 256, 0, stream>>>(prev16, cur16, Mn * 256);

    for (int s = 0; s < 15; ++s) {
      const size_t qbase = (size_t)(s + 1) * Mn;
      const int* ridx = preds + (size_t)s * Mn * 16;

      // Q = prev[nodes] @ Wq^T + bq
      gemm_wmma_kernel<256, false, false, true, false>
          <<<dim3(4, Mn / 64), 128, 0, stream>>>(
              prev16 + qbase * 256, nullptr,
              wq16 + (size_t)l * 65536, bq + l * 256, nullptr, nullptr,
              Q16, 256, nullptr, 0, nullptr, 0);

      // K,V = cur[preds] @ {Wk,Wv}^T + {bk,bv}  (gathered A shared by both)
      gemm_wmma_kernel<256, true, true, true, false>
          <<<dim3(4, N / 64), 128, 0, stream>>>(
              cur16, ridx,
              wk16 + (size_t)l * 65536, bk + l * 256,
              wv16 + (size_t)l * 65536, bv + l * 256,
              K16, 256, nullptr, 0, V16, 256);

      // attention + q copy into concat front half
      attn_kernel<<<512, 256, 0, stream>>>(Q16, K16, V16,
                                           prev16 + qbase * 256, agg16, cat16);

      // O = agg @ Wo^T + bo  -> concat back half
      gemm_wmma_kernel<256, false, false, true, false>
          <<<dim3(4, Mn / 64), 128, 0, stream>>>(
              agg16, nullptr,
              wo16 + (size_t)l * 65536, bo + l * 256, nullptr, nullptr,
              cat16 + 256, 512, nullptr, 0, nullptr, 0);

      // z = cat @ Wc^T + bc   (K = 512, f32 out for LN precision)
      gemm_wmma_kernel<512, false, false, false, true>
          <<<dim3(4, Mn / 64), 128, 0, stream>>>(
              cat16, nullptr,
              wc16 + (size_t)l * 131072, bc + l * 256, nullptr, nullptr,
              nullptr, 0, z32, 256, nullptr, 0);

      // LayerNorm + GELU -> d_out (f32) and cur16 (f16)
      ln_gelu_kernel<<<512, 256, 0, stream>>>(z32, ln_g + l * 256, ln_b + l * 256,
                                              out + qbase * 1024 + (size_t)(l + 1) * 256,
                                              cur16 + qbase * 256);
    }
    _Float16* t = prev16; prev16 = cur16; cur16 = t;
  }
}
